// SelfAttention_17368847745340
// MI455X (gfx1250) — matrix operand order repaired and checked by reading, not compiled, not run on hardware
//
#include <hip/hip_runtime.h>
#include <hip/hip_bf16.h>

typedef __attribute__((ext_vector_type(16))) _Float16 v16h;
typedef __attribute__((ext_vector_type(8)))  _Float16 v8h;
typedef __attribute__((ext_vector_type(8)))  float    v8f;
typedef __attribute__((ext_vector_type(4)))  unsigned int u32x4;
typedef __attribute__((ext_vector_type(8)))  int i32x8;
typedef __attribute__((ext_vector_type(4)))  int i32x4;

#define WMMA_F16(a, b, c) \
  __builtin_amdgcn_wmma_f32_16x16x32_f16(false, (a), false, (b), (short)0, (c), false, false)

#define BATCH 4
#define SEQ   4096
#define DIM   256

#if defined(__AMDGCN__) && __has_builtin(__builtin_amdgcn_tensor_load_to_lds)
#define HAVE_TDM 1
#else
#define HAVE_TDM 0
#endif

__device__ __forceinline__ v16h cat16(v8h lo, v8h hi) {
  v16h r;
#pragma unroll
  for (int i = 0; i < 8; ++i) { r[i] = lo[i]; r[i + 8] = hi[i]; }
  return r;
}

// A-fragment (16x32 f16): lane M = lane%16; 16 contiguous k values
// starting at k0 + 16*hiLane (matches B-fragment k order).
__device__ __forceinline__ v16h load_a_frag(const _Float16* base, int stride,
                                            int rowBase, int k0, int laneN, int hi) {
  const _Float16* p = base + (size_t)(rowBase + laneN) * stride + k0 + 16 * hi;
  v8h lo = *(const v8h*)(p);
  v8h hh = *(const v8h*)(p + 8);
  return cat16(lo, hh);
}

// B-fragment (32x16 f16), B[k][n] with n = lane%16; 16 contiguous k values
// starting at k0 + 16*hiLane, gathered from an n-major (row = n) buffer.
__device__ __forceinline__ v16h load_b_frag(const _Float16* base, int stride,
                                            int nBase, int k0, int laneN, int hi) {
  const _Float16* p = base + (size_t)(nBase + laneN) * stride + k0 + 16 * hi;
  v8h lo = *(const v8h*)(p);
  v8h hh = *(const v8h*)(p + 8);
  return cat16(lo, hh);
}

#if HAVE_TDM
// Issue one TDM 2-D tile load (Global -> LDS), f16 elements (data_size=1 -> 2B).
// Descriptor layout per CDNA5 ISA 8.3/8.4: group0 {count=1, lds_addr,
// global_addr, type=2}; group1 {data_size, tensor_dim0/1, tile_dim0/1,
// tensor_dim0_stride}.  groups 2/3 zero (2-D tensor).  This toolchain exposes
// the 6-arg builtin form (extra i32x8 group, zero-filled).
__device__ __forceinline__ void tdm_load_2d(unsigned lds_addr, const void* gptr,
                                            unsigned tensor_d0, unsigned tensor_d1,
                                            unsigned tile_d0, unsigned tile_d1,
                                            unsigned long long stride0) {
  unsigned long long ga = (unsigned long long)gptr;
  u32x4 g0;
  g0[0] = 1u;                                                  // count=1, user mode
  g0[1] = lds_addr;                                            // LDS byte address
  g0[2] = (unsigned)ga;                                        // global_addr[31:0]
  g0[3] = (unsigned)((ga >> 32) & 0x01FFFFFFu) | (2u << 30);   // addr[56:32] | type=2
  i32x8 g1;
  g1[0] = (int)(1u << 16);                                     // data_size=1 (2 bytes)
  g1[1] = (int)((tensor_d0 & 0xFFFFu) << 16);                  // tensor_dim0[15:0]
  g1[2] = (int)((tensor_d0 >> 16) | ((tensor_d1 & 0xFFFFu) << 16));
  g1[3] = (int)((tensor_d1 >> 16) | ((tile_d0 & 0xFFFFu) << 16));
  g1[4] = (int)(tile_d1 & 0xFFFFu);                            // tile_dim1, tile_dim2=0
  g1[5] = (int)(unsigned)(stride0 & 0xFFFFFFFFull);            // dim0_stride[31:0]
  g1[6] = (int)(unsigned)(stride0 >> 32);                      // dim0_stride[47:32]
  g1[7] = 0;
  i32x4 z4; z4[0] = 0; z4[1] = 0; z4[2] = 0; z4[3] = 0;
  i32x8 z8;
#pragma unroll
  for (int i = 0; i < 8; ++i) z8[i] = 0;
  __builtin_amdgcn_tensor_load_to_lds(g0, g1, z4, z4, z8, 0);
}

__device__ __forceinline__ void tdm_wait0() {
#if __has_builtin(__builtin_amdgcn_s_wait_tensorcnt)
  __builtin_amdgcn_s_wait_tensorcnt(0);
#else
  asm volatile("s_wait_tensorcnt 0" ::: "memory");
#endif
}
#endif  // HAVE_TDM

// ---------------------------------------------------------------------------
// Kernel 1: fused QKV projection.  y = x @ W^T + b, f16 WMMA, f32 accum.
// Q is pre-scaled by 1/sqrt(256); V is written transposed [B, dim, key].
// ---------------------------------------------------------------------------
__device__ __forceinline__ void proj_one(
    const _Float16* xs, const float* __restrict__ W, const float* __restrict__ bias,
    float scale, int r0, int rBase, int colHalf, int laneN, int hi,
    _Float16* __restrict__ out, int transposedV) {
  v8f acc[8];
#pragma unroll
  for (int t = 0; t < 8; ++t)
#pragma unroll
    for (int e = 0; e < 8; ++e) acc[t][e] = 0.0f;

#pragma unroll
  for (int kk = 0; kk < 8; ++kk) {
    const int k0 = kk * 32;
    v16h a = load_a_frag(xs, DIM, rBase, k0, laneN, hi);
#pragma unroll
    for (int t = 0; t < 8; ++t) {
      const int n0 = colHalf * 128 + t * 16;
      const float* wp = W + (size_t)(n0 + laneN) * DIM + k0 + 16 * hi;
      v16h b;
#pragma unroll
      for (int q = 0; q < 16; q += 4) {
        float4 f = *(const float4*)(wp + q);
        b[q + 0] = (_Float16)f.x; b[q + 1] = (_Float16)f.y;
        b[q + 2] = (_Float16)f.z; b[q + 3] = (_Float16)f.w;
      }
      acc[t] = WMMA_F16(a, b, acc[t]);
    }
  }
  // C/D layout: element (M = v + 8*hiLane, N = laneN)
#pragma unroll
  for (int t = 0; t < 8; ++t) {
    const int col = colHalf * 128 + t * 16 + laneN;
    const float bc = bias[col];
#pragma unroll
    for (int v = 0; v < 8; ++v) {
      const int row = r0 + rBase + v + 8 * hi;      // global row in [0, B*S)
      const float val = (acc[t][v] + bc) * scale;
      if (!transposedV) {
        out[(size_t)row * DIM + col] = (_Float16)val;
      } else {
        const int bb = row >> 12, srow = row & (SEQ - 1);
        out[((size_t)bb * DIM + col) * SEQ + srow] = (_Float16)val;
      }
    }
  }
}

__global__ __launch_bounds__(256) void qkv_kernel(
    const float* __restrict__ x,
    const float* __restrict__ Wq, const float* __restrict__ bq,
    const float* __restrict__ Wk, const float* __restrict__ bk,
    const float* __restrict__ Wv, const float* __restrict__ bv,
    _Float16* __restrict__ Qh, _Float16* __restrict__ Kh, _Float16* __restrict__ Vt) {
  __shared__ _Float16 xs[64 * DIM];            // 32 KB
  const int tid = threadIdx.x;
  const int r0 = blockIdx.x * 64;
  // stage x tile -> f16 LDS, coalesced float4 loads
#pragma unroll
  for (int i = 0; i < 16; ++i) {
    const int linear = i * 256 + tid;          // 4096 float4 chunks
    const int row = linear >> 6;
    const int c4 = (linear & 63) * 4;
    float4 f = *(const float4*)(x + (size_t)(r0 + row) * DIM + c4);
    _Float16* d = xs + row * DIM + c4;
    d[0] = (_Float16)f.x; d[1] = (_Float16)f.y;
    d[2] = (_Float16)f.z; d[3] = (_Float16)f.w;
  }
  __syncthreads();

  const int wave = tid >> 5, lane = tid & 31;
  const int laneN = lane & 15, hi = lane >> 4;
  const int rBase = (wave >> 1) * 16;          // 4 row tiles
  const int colHalf = wave & 1;                // 2 column halves

  proj_one(xs, Wq, bq, 0.0625f, r0, rBase, colHalf, laneN, hi, Qh, 0);  // 1/sqrt(256)
  proj_one(xs, Wk, bk, 1.0f,    r0, rBase, colHalf, laneN, hi, Kh, 0);
  proj_one(xs, Wv, bv, 1.0f,    r0, rBase, colHalf, laneN, hi, Vt, 1);
}

// ---------------------------------------------------------------------------
// Kernel 2: flash attention.  64 query rows / block, 4 waves, each wave owns
// 16 rows (full online softmax in-wave).  K/V tiles double-buffered in LDS,
// fetched by the Tensor Data Mover: tile j+1 DMA overlaps WMMA on tile j.
// ---------------------------------------------------------------------------
__global__ __launch_bounds__(128) void attn_kernel(
    const _Float16* __restrict__ Qh, const _Float16* __restrict__ Kh,
    const _Float16* __restrict__ Vt, float* __restrict__ out) {
  __shared__ _Float16 Qs[64 * DIM];            // 32 KB
  __shared__ _Float16 Ks[2][64 * DIM];         // 64 KB [key][dim], double buffered
  __shared__ _Float16 Vs[2][DIM * 64];         // 64 KB [dim][key], double buffered
  __shared__ _Float16 Ps[4 * 16 * 64];         //  8 KB wave-private P scratch

  const int tid = threadIdx.x;
  const int wave = tid >> 5, lane = tid & 31;
  const int laneN = lane & 15, hi = lane >> 4;
  const int bb = blockIdx.y;
  const size_t qrow0 = (size_t)bb * SEQ + blockIdx.x * 64;

  // stage Q tile (already pre-scaled)
#pragma unroll
  for (int i = 0; i < 16; ++i) {
    const int chunk = i * 128 + tid;           // 2048 chunks of 8 halves
    const int row = chunk >> 5, c8 = (chunk & 31) * 8;
    *(v8h*)(Qs + row * DIM + c8) = *(const v8h*)(Qh + (qrow0 + row) * DIM + c8);
  }
  __syncthreads();

  float m_i[8], l_i[8];
#pragma unroll
  for (int v = 0; v < 8; ++v) { m_i[v] = -1e30f; l_i[v] = 0.0f; }
  v8f o[16];
#pragma unroll
  for (int n = 0; n < 16; ++n)
#pragma unroll
    for (int e = 0; e < 8; ++e) o[n][e] = 0.0f;

  _Float16* Pw = Ps + wave * 16 * 64;
  const int rBase = wave * 16;

#if HAVE_TDM
  // Prologue: DMA tile 0 into buffer 0 (K: contiguous rows; V: strided slice).
  if (wave == 0) {
    tdm_load_2d((unsigned)(size_t)&Ks[0][0], Kh + ((size_t)bb * SEQ + 0) * DIM,
                DIM, SEQ, DIM, 64, DIM);
    tdm_load_2d((unsigned)(size_t)&Vs[0][0], Vt + (size_t)bb * DIM * SEQ + 0,
                SEQ, DIM, 64, DIM, SEQ);
  }
#endif

  for (int kt = 0; kt < SEQ / 64; ++kt) {
    const int kb = kt * 64;
    const int cur = kt & 1;

#if HAVE_TDM
    if (wave == 0) tdm_wait0();                // buffer `cur` now holds tile kt
    __syncthreads();                           // publish; also: all waves done with buf cur^1
    if (kt + 1 < SEQ / 64 && wave == 0) {      // DMA tile kt+1 under compute of tile kt
      const int kn = kb + 64;
      tdm_load_2d((unsigned)(size_t)&Ks[cur ^ 1][0],
                  Kh + ((size_t)bb * SEQ + kn) * DIM, DIM, SEQ, DIM, 64, DIM);
      tdm_load_2d((unsigned)(size_t)&Vs[cur ^ 1][0],
                  Vt + (size_t)bb * DIM * SEQ + kn, SEQ, DIM, 64, DIM, SEQ);
    }
#else
    __syncthreads();
    // cooperative staging fallback (no TDM builtin on this toolchain)
#pragma unroll
    for (int i = 0; i < 16; ++i) {
      const int chunk = i * 128 + tid;
      const int row = chunk >> 5, c8 = (chunk & 31) * 8;
      *(v8h*)(&Ks[cur][0] + row * DIM + c8) =
          *(const v8h*)(Kh + ((size_t)bb * SEQ + kb + row) * DIM + c8);
    }
#pragma unroll
    for (int i = 0; i < 16; ++i) {
      const int chunk = i * 128 + tid;
      const int dim = chunk >> 3, k8 = (chunk & 7) * 8;
      *(v8h*)(&Vs[cur][0] + dim * 64 + k8) =
          *(const v8h*)(Vt + ((size_t)bb * DIM + dim) * SEQ + kb + k8);
    }
    __syncthreads();
#endif

    // S = Q K^T  (16 x 64 per wave, f32 accum)
    v8f s[4];
#pragma unroll
    for (int t = 0; t < 4; ++t)
#pragma unroll
      for (int e = 0; e < 8; ++e) s[t][e] = 0.0f;
#pragma unroll
    for (int kk = 0; kk < 8; ++kk) {
      v16h a = load_a_frag(Qs, DIM, rBase, kk * 32, laneN, hi);
#pragma unroll
      for (int t = 0; t < 4; ++t) {
        v16h bfr = load_b_frag(&Ks[cur][0], DIM, t * 16, kk * 32, laneN, hi);
        s[t] = WMMA_F16(a, bfr, s[t]);
      }
    }

    // online softmax: rows live at (v, hiLane); reduce across 16 N-lanes
    float alpha[8];
#pragma unroll
    for (int v = 0; v < 8; ++v) {
      float rmax = fmaxf(fmaxf(s[0][v], s[1][v]), fmaxf(s[2][v], s[3][v]));
#pragma unroll
      for (int mm = 1; mm < 16; mm <<= 1) rmax = fmaxf(rmax, __shfl_xor(rmax, mm, 32));
      const float mnew = fmaxf(m_i[v], rmax);
      alpha[v] = __expf(m_i[v] - mnew);
      m_i[v] = mnew;
      float rs = 0.0f;
#pragma unroll
      for (int t = 0; t < 4; ++t) {
        const float p = __expf(s[t][v] - mnew);
        rs += p;
        Pw[(v + 8 * hi) * 64 + t * 16 + laneN] = (_Float16)p;   // C-layout -> LDS
      }
#pragma unroll
      for (int mm = 1; mm < 16; mm <<= 1) rs += __shfl_xor(rs, mm, 32);
      l_i[v] = l_i[v] * alpha[v] + rs;
    }
    asm volatile("s_wait_dscnt 0" ::: "memory");  // P stores visible to A-frag loads

    // rescale running O, then O += P V
#pragma unroll
    for (int n = 0; n < 16; ++n)
#pragma unroll
      for (int v = 0; v < 8; ++v) o[n][v] *= alpha[v];
#pragma unroll
    for (int ks = 0; ks < 2; ++ks) {
      v16h a = load_a_frag(Pw, 64, 0, ks * 32, laneN, hi);
#pragma unroll
      for (int n = 0; n < 16; ++n) {
        v16h bfr = load_b_frag(&Vs[cur][0], 64, n * 16, ks * 32, laneN, hi);
        o[n] = WMMA_F16(a, bfr, o[n]);
      }
    }
  }

  // normalize and write fp32 output
#pragma unroll
  for (int v = 0; v < 8; ++v) {
    const float inv = 1.0f / l_i[v];
    const size_t row = qrow0 + rBase + v + 8 * hi;
#pragma unroll
    for (int n = 0; n < 16; ++n)
      out[row * DIM + n * 16 + laneN] = o[n][v] * inv;
  }
}

extern "C" void kernel_launch(void* const* d_in, const int* in_sizes, int n_in,
                              void* d_out, int out_size, void* d_ws, size_t ws_size,
                              hipStream_t stream) {
  const float* x  = (const float*)d_in[0];
  const float* Wq = (const float*)d_in[1];
  const float* bq = (const float*)d_in[2];
  const float* Wk = (const float*)d_in[3];
  const float* bk = (const float*)d_in[4];
  const float* Wv = (const float*)d_in[5];
  const float* bv = (const float*)d_in[6];
  float* out = (float*)d_out;

  char* ws = (char*)d_ws;
  const size_t MAT_BYTES = (size_t)BATCH * SEQ * DIM * sizeof(_Float16);  // 8 MB
  _Float16* Qh = (_Float16*)(ws);
  _Float16* Kh = (_Float16*)(ws + MAT_BYTES);
  _Float16* Vt = (_Float16*)(ws + 2 * MAT_BYTES);

  qkv_kernel<<<(BATCH * SEQ) / 64, 256, 0, stream>>>(x, Wq, bq, Wk, bk, Wv, bv, Qh, Kh, Vt);
  attn_kernel<<<dim3(SEQ / 64, BATCH), 128, 0, stream>>>(Qh, Kh, Vt, out);
}
